// TokenAlign_86105504350469
// MI455X (gfx1250) — matrix operand order repaired and checked by reading, not compiled, not run on hardware
//
#include <hip/hip_runtime.h>
#include <math.h>

#define Bb 8
#define Nn 1024
#define Mm 2048
#define Dd 1024

typedef __bf16 v16bf __attribute__((ext_vector_type(16)));
typedef float  v8f   __attribute__((ext_vector_type(8)));
typedef float  v4f   __attribute__((ext_vector_type(4)));
typedef unsigned int   v8u  __attribute__((ext_vector_type(8)));
typedef unsigned short v8us __attribute__((ext_vector_type(8)));
typedef unsigned short v4us __attribute__((ext_vector_type(4)));

// fp32 -> bf16 hi (truncation) + bf16 lo (exact residual, truncated).
// a*b ~= ah*bh + al*bh + ah*bl with fp32 accumulate: ~2^-16 relative error.
__device__ __forceinline__ void splitf(float f, unsigned short& h, unsigned short& l) {
    unsigned u = __float_as_uint(f);
    h = (unsigned short)(u >> 16);
    float hf = __uint_as_float(u & 0xFFFF0000u);
    l = (unsigned short)(__float_as_uint(f - hf) >> 16);
}

// Pack 16 floats into hi/lo v16bf using v_perm_b32 (1 op per packed pair for hi).
__device__ __forceinline__ void pack16(const float* f, v16bf& hi, v16bf& lo) {
    v8u uh, ul;
#pragma unroll
    for (int i = 0; i < 8; ++i) {
        unsigned a = __float_as_uint(f[2 * i]);
        unsigned b = __float_as_uint(f[2 * i + 1]);
        uh[i] = __builtin_amdgcn_perm(b, a, 0x07060302u);   // {b.hi16, a.hi16}
        float fa = __uint_as_float(a & 0xFFFF0000u);
        float fb = __uint_as_float(b & 0xFFFF0000u);
        unsigned la = __float_as_uint(f[2 * i] - fa);
        unsigned lb = __float_as_uint(f[2 * i + 1] - fb);
        ul[i] = __builtin_amdgcn_perm(lb, la, 0x07060302u);
    }
    hi = __builtin_bit_cast(v16bf, uh);
    lo = __builtin_bit_cast(v16bf, ul);
}

union Frag {  // two 16B LDS loads -> one WMMA operand
    v16bf v;
    v8us  h[2];
};

__device__ __forceinline__ v8f wmma3(v8f acc, v16bf ah, v16bf al, v16bf bh, v16bf bl) {
    acc = __builtin_amdgcn_wmma_f32_16x16x32_bf16(false, ah, false, bh, (short)0, acc, false, false);
    acc = __builtin_amdgcn_wmma_f32_16x16x32_bf16(false, al, false, bh, (short)0, acc, false, false);
    acc = __builtin_amdgcn_wmma_f32_16x16x32_bf16(false, ah, false, bl, (short)0, acc, false, false);
    return acc;
}

__launch_bounds__(256, 1)
__global__ void tokenalign_fused(const float* __restrict__ text,
                                 const float* __restrict__ audio,
                                 const float* __restrict__ sigma,
                                 const int*   __restrict__ lengths,
                                 float* __restrict__ enhanced,
                                 float* __restrict__ attn) {
    // 256 KB LDS, time-multiplexed (WGP budget is 320 KB):
    //   [0,128K)   : S fp32 16x2048 score tile            (phases 1-2)
    //   [128K,192K): tHi/tLo bf16 text planes 16x1024     (phase 1)
    //   [128K,256K): pHi/pLo bf16 probability planes 16x2048 (phases 2-3, overlays tHi/tLo)
    __shared__ __align__(16) unsigned char ldsraw[256 * 1024];
    float (*S)[Mm]      = (float (*)[Mm])(ldsraw);
    unsigned short* tHi = (unsigned short*)(ldsraw + 128 * 1024);
    unsigned short* tLo = (unsigned short*)(ldsraw + 160 * 1024);
    unsigned short* pHi = (unsigned short*)(ldsraw + 128 * 1024);
    unsigned short* pLo = (unsigned short*)(ldsraw + 192 * 1024);

    const int wg   = blockIdx.x;
    const int b    = wg >> 6;          // / (Nn/16)
    const int nt   = wg & 63;
    const int row0 = nt * 16;

    const int tid    = threadIdx.x;
    const int wave   = tid >> 5;
    const int lane   = tid & 31;
    const int lane15 = lane & 15;
    const int half   = lane >> 4;      // 0 or 1

    const float sg     = fabsf(sigma[0]) + 1e-8f;
    const float inv2s2 = 0.5f / (sg * sg);
    const int   len    = lengths[b];
    const float scale  = 0.03125f;     // 1/sqrt(1024)

    // ---------- Phase 0: split text tile (16 x D) into bf16 hi/lo planes in LDS ----------
#pragma unroll 4
    for (int i = tid; i < 16 * Dd / 4; i += 256) {
        const int r  = i >> 8;                 // Dd/4 = 256 quads per row
        const int k4 = (i & 255) * 4;
        v4f f = *(const v4f*)(text + ((size_t)b * Nn + row0 + r) * (size_t)Dd + k4);
        v4us h, l;
#pragma unroll
        for (int j = 0; j < 4; ++j) {
            unsigned short hh, ll;
            splitf(f[j], hh, ll);
            h[j] = hh;
            l[j] = ll;
        }
        *(v4us*)(tHi + (r << 10) + k4) = h;
        *(v4us*)(tLo + (r << 10) + k4) = l;
    }
    __syncthreads();

    // ---------- Phase 1: S = scale*Q.K^T + bias, masked ----------
    // A (16x32 bf16): lanes<16 row=lane, K 0..15; lanes>=16 K 16..31 -> ds_load_b128 x2
    // B (32x16 bf16): lanes<16 col=lane, K 0..15; lanes>=16 K 16..31 (contiguous since B = audio^T)
#pragma unroll 1
    for (int t = 0; t < 16; ++t) {
        const int ct = wave * 16 + t;          // audio column tile (16 audio rows)
        const float* aB = audio + ((size_t)b * Mm + (size_t)ct * 16 + lane15) * (size_t)Dd;
        v8f acc = {0.f, 0.f, 0.f, 0.f, 0.f, 0.f, 0.f, 0.f};
#pragma unroll 4
        for (int kc = 0; kc < Dd / 32; ++kc) {
            const int ka = kc * 32 + half * 16;
            Frag ah, al;
            ah.h[0] = *(const v8us*)(tHi + (lane15 << 10) + ka);
            ah.h[1] = *(const v8us*)(tHi + (lane15 << 10) + ka + 8);
            al.h[0] = *(const v8us*)(tLo + (lane15 << 10) + ka);
            al.h[1] = *(const v8us*)(tLo + (lane15 << 10) + ka + 8);

            float fb[16];
            const int kb = kc * 32 + half * 16;
            *(v4f*)&fb[0]  = *(const v4f*)(aB + kb);
            *(v4f*)&fb[4]  = *(const v4f*)(aB + kb + 4);
            *(v4f*)&fb[8]  = *(const v4f*)(aB + kb + 8);
            *(v4f*)&fb[12] = *(const v4f*)(aB + kb + 12);
            v16bf bh, bl;
            pack16(fb, bh, bl);
            acc = wmma3(acc, ah.v, al.v, bh, bl);
        }
        // epilogue: scale + Gaussian positional bias + length mask -> LDS
        const int col = ct * 16 + lane15;
        const float ap = (float)col * (1.0f / (float)Mm);
#pragma unroll
        for (int v = 0; v < 8; ++v) {
            const int rloc = v + half * 8;
            const float tp = (float)(row0 + rloc) * (1.0f / (float)Nn);
            const float dr = tp - ap;
            float s = acc[v] * scale - dr * dr * inv2s2;
            if (col >= len) s = -__builtin_inff();
            S[rloc][col] = s;
        }
    }
    __syncthreads();

    // ---------- Phase 2: row softmax over M; emit attention + split P into LDS bf16 planes ----------
#pragma unroll 1
    for (int rr = 0; rr < 2; ++rr) {
        const int r = wave * 2 + rr;
        float m = -__builtin_inff();
        for (int c = lane; c < Mm; c += 32) m = fmaxf(m, S[r][c]);
#pragma unroll
        for (int off = 16; off > 0; off >>= 1) m = fmaxf(m, __shfl_xor(m, off, 32));
        float sum = 0.0f;
        for (int c = lane; c < Mm; c += 32) {
            float e = __expf(S[r][c] - m);
            S[r][c] = e;
            sum += e;
        }
#pragma unroll
        for (int off = 16; off > 0; off >>= 1) sum += __shfl_xor(sum, off, 32);
        const float inv = 1.0f / sum;
        float* arow = attn + ((size_t)b * Nn + row0 + r) * (size_t)Mm;
        for (int c = lane; c < Mm; c += 32) {
            float p = S[r][c] * inv;
            arow[c] = p;                       // coalesced attention output
            unsigned short hh, ll;
            splitf(p, hh, ll);
            pHi[(r << 11) + c] = hh;           // P planes for phase 3 (overlay region)
            pLo[(r << 11) + c] = ll;
        }
    }
    __syncthreads();

    // ---------- Phase 3: enhanced = P @ audio ----------
    // A = P from LDS bf16 planes (ds_load_b128); B = audio (M x D): per-lane strided over M,
    // coalesced across lanes in D, split inline (each audio element converted once per WG).
#pragma unroll 1
    for (int t = 0; t < 8; ++t) {
        const int dt   = wave * 8 + t;         // D column tile
        const int dcol = dt * 16 + lane15;
        const float* pB = audio + (size_t)b * Mm * Dd + dcol;
        v8f acc = {0.f, 0.f, 0.f, 0.f, 0.f, 0.f, 0.f, 0.f};
#pragma unroll 2
        for (int mc = 0; mc < Mm / 32; ++mc) {
            const int ka = mc * 32 + half * 16;
            Frag ah, al;
            ah.h[0] = *(const v8us*)(pHi + (lane15 << 11) + ka);
            ah.h[1] = *(const v8us*)(pHi + (lane15 << 11) + ka + 8);
            al.h[0] = *(const v8us*)(pLo + (lane15 << 11) + ka);
            al.h[1] = *(const v8us*)(pLo + (lane15 << 11) + ka + 8);

            float fb[16];
            const int kb = mc * 32 + half * 16;
#pragma unroll
            for (int k = 0; k < 16; ++k)
                fb[k] = pB[(size_t)(kb + k) * Dd];
            v16bf bh, bl;
            pack16(fb, bh, bl);
            acc = wmma3(acc, ah.v, al.v, bh, bl);
        }
#pragma unroll
        for (int v = 0; v < 8; ++v) {
            const int rloc = v + half * 8;
            enhanced[((size_t)b * Nn + row0 + rloc) * (size_t)Dd + dcol] = acc[v];
        }
    }
}

extern "C" void kernel_launch(void* const* d_in, const int* in_sizes, int n_in,
                              void* d_out, int out_size, void* d_ws, size_t ws_size,
                              hipStream_t stream) {
    const float* text  = (const float*)d_in[0];
    const float* audio = (const float*)d_in[1];
    const float* sigma = (const float*)d_in[2];
    const int*   lens  = (const int*)d_in[3];
    float* enhanced = (float*)d_out;
    float* attn     = (float*)d_out + (size_t)Bb * Nn * Dd;

    dim3 grid(Bb * (Nn / 16));   // 512 workgroups: one per (batch, 16 text rows)
    dim3 block(256);             // 8 wave32 waves
    hipLaunchKernelGGL(tokenalign_fused, grid, block, 0, stream,
                       text, audio, sigma, lens, enhanced, attn);
}
